// DQA_89077621719347
// MI455X (gfx1250) — compile-verified, matrix operand order checked by reading, not run on hardware
//
#include <hip/hip_runtime.h>
#include <hip/hip_fp16.h>

#define B_   16
#define C_   64
#define H_   256
#define W_   256
#define HW_  (H_*W_)

typedef __attribute__((ext_vector_type(16))) _Float16 v16h;
typedef __attribute__((ext_vector_type(8)))  float    v8f;

union Frag16 { v16h v; unsigned int u[8]; };

__device__ __forceinline__ float lrelu01(float x) { return x > 0.f ? x : 0.1f * x; }

// ---------------- kernel 1: per-(b,c) spatial mean ----------------
__global__ __launch_bounds__(256) void dqa_mean(const float* __restrict__ feat,
                                                float* __restrict__ fa) {
  int idx = blockIdx.x;  // 0..B*C-1
  const float4* p = (const float4*)(feat + (size_t)idx * HW_);
  float s = 0.f;
  for (int i = threadIdx.x; i < HW_ / 4; i += 256) {
    float4 t = p[i];
    s += (t.x + t.y) + (t.z + t.w);
  }
#pragma unroll
  for (int off = 16; off > 0; off >>= 1) s += __shfl_down(s, off, 32);
  __shared__ float red[8];
  if ((threadIdx.x & 31) == 0) red[threadIdx.x >> 5] = s;
  __syncthreads();
  if (threadIdx.x == 0) {
    float t = 0.f;
#pragma unroll
    for (int i = 0; i < 8; ++i) t += red[i];
    fa[idx] = t * (1.0f / (float)HW_);
  }
}

// ---------------- kernel 2: scalar chain (all tiny GEMMs) ----------------
__global__ __launch_bounds__(256) void dqa_prep(
    const float* __restrict__ fa, const float* __restrict__ deg,
    const float* __restrict__ wq, const float* __restrict__ bq,
    const float* __restrict__ wk, const float* __restrict__ bk,
    const float* __restrict__ wv, const float* __restrict__ bv,
    const float* __restrict__ gamma, const float* __restrict__ k_w1,
    const float* __restrict__ k_w2, const float* __restrict__ conv_w,
    const float* __restrict__ ca_w1, const float* __restrict__ ca_w2,
    float* __restrict__ kern_out, float* __restrict__ att_out,
    __half* __restrict__ wh_out) {
  __shared__ float q[B_*C_], kk[B_*C_], v[B_*C_], emb[B_*C_], hid[B_*C_], h8[B_*8];
  const int tid = threadIdx.x;

  for (int idx = tid; idx < B_*C_; idx += 256) {
    int b = idx >> 6, j = idx & 63;
    float sq = bq[j], sk = bk[j], sv = bv[j];
    for (int c = 0; c < C_; ++c) {
      float fc = fa[b*C_ + c], dc = deg[b*C_ + c];
      sq += fc * wq[j*C_ + c];
      sk += dc * wk[j*C_ + c];
      sv += dc * wv[j*C_ + c];
    }
    q[idx] = sq; kk[idx] = sk; v[idx] = sv;
  }
  __syncthreads();

  const float g = gamma[0];
  for (int idx = tid; idx < B_*C_; idx += 256) {
    int b = idx >> 6;
    float qv = q[idx];
    float m = -3.4e38f;
    for (int i = 0; i < C_; ++i) m = fmaxf(m, qv * kk[b*C_ + i]);
    float s = 0.f, d = 0.f;
    for (int i = 0; i < C_; ++i) {
      float e = __expf(qv * kk[b*C_ + i] - m);
      s += e; d += v[b*C_ + i] * e;
    }
    emb[idx] = g * (d / s) + deg[idx];
  }
  __syncthreads();

  for (int idx = tid; idx < B_*C_; idx += 256) {
    int b = idx >> 6, j = idx & 63;
    float h = 0.f;
    for (int c = 0; c < C_; ++c) h += emb[b*C_ + c] * k_w1[j*C_ + c];
    hid[idx] = lrelu01(h);
  }
  if (tid < B_ * 8) {
    int b = tid >> 3, r = tid & 7;
    float h = 0.f;
    for (int c = 0; c < C_; ++c) h += deg[b*C_ + c] * ca_w1[r*C_ + c];
    h8[tid] = lrelu01(h);
  }
  __syncthreads();

  for (int idx = tid; idx < B_ * 576; idx += 256) {       // per-(b,c) 3x3 kernels
    int b = idx / 576, t = idx - b * 576;
    float s = 0.f;
    for (int c = 0; c < C_; ++c) s += hid[b*C_ + c] * k_w2[t*C_ + c];
    kern_out[idx] = s;
  }
  for (int idx = tid; idx < B_*C_; idx += 256) {          // channel attention
    int b = idx >> 6, c = idx & 63;
    float s = 0.f;
#pragma unroll
    for (int r = 0; r < 8; ++r) s += h8[b*8 + r] * ca_w2[c*8 + r];
    att_out[idx] = 1.0f / (1.0f + __expf(-s));
  }
  for (int i = tid; i < C_*C_; i += 256) wh_out[i] = __float2half(conv_w[i]);
}

// ---------------- kernel 3: fused depthwise + 1x1(WMMA) + residual ----------------
// Tile: 32 wide x 8 tall (256 pixels). Halo tile per channel: 10 rows x 34 cols.
#define TW_ 32
#define TH_ 8
#define HLW_ 34                              // halo row width
#define FT_STRIDE (HLW_*10)                  // 340 floats per channel
#define X_STRIDE  33                         // dwords per pixel (64ch f16 + pad) -> odd, conflict-free
#define FT_OFF    0
#define X_OFF     (C_*FT_STRIDE*4)           // 87040
#define KN_OFF    (X_OFF + 256*X_STRIDE*4)   // 120832
#define WS_OFF    (KN_OFF + 576*4)           // 123136
#define WS_STRIDE 66                         // halfs per outch row (33 dwords)
#define AT_OFF    (WS_OFF + C_*WS_STRIDE*2)  // 131584
#define CB_OFF    (AT_OFF + 256)             // 131840
#define SMEM_BYTES (CB_OFF + 256)            // 132096 bytes (gfx1250: up to 320KB/WG)

__global__ __launch_bounds__(256) void dqa_main(
    const float* __restrict__ feat, const float* __restrict__ conv_b,
    const float* __restrict__ kern, const float* __restrict__ att,
    const __half* __restrict__ wh, float* __restrict__ out) {
  extern __shared__ char smem[];
  float*  ft   = (float*)(smem + FT_OFF);
  __half* xh   = (__half*)(smem + X_OFF);
  float*  kn   = (float*)(smem + KN_OFF);
  __half* wsh  = (__half*)(smem + WS_OFF);
  float*  attb = (float*)(smem + AT_OFF);
  float*  cb   = (float*)(smem + CB_OFF);

  const int tid = threadIdx.x;
  const int b  = blockIdx.z;
  const int h0 = blockIdx.y * TH_;
  const int w0 = blockIdx.x * TW_;
  const float* fb = feat + (size_t)b * C_ * HW_;

  // Phase 1: stage 10x34x64 fp32 halo tile (zero padded); rows are 136B
  // contiguous runs in global -> good coalescing.
  for (int i = tid; i < C_ * FT_STRIDE; i += 256) {
    int c = i / FT_STRIDE, rem = i - c * FT_STRIDE;
    int r = rem / HLW_, cc = rem - r * HLW_;
    int h = h0 + r - 1, w = w0 + cc - 1;
    float val = 0.f;
    if (h >= 0 && h < H_ && w >= 0 && w < W_) val = fb[c*HW_ + h*W_ + w];
    ft[c*FT_STRIDE + rem] = val;
  }
  for (int i = tid; i < 576; i += 256) kn[i] = kern[b*576 + i];
  for (int i = tid; i < C_*C_; i += 256) {
    int o = i >> 6, c = i & 63;
    wsh[o*WS_STRIDE + c] = wh[i];
  }
  if (tid < C_) { attb[tid] = att[b*C_ + tid]; cb[tid] = conv_b[tid]; }
  __syncthreads();

  // Phase 2: dynamic depthwise 3x3 + leaky relu -> f16 LDS.
  // Each thread owns a 4-pixel horizontal strip per channel: 18 LDS reads
  // (3 rows x 6 floats, pairable into ds_load_2addr) produce 4 outputs.
  {
    const int strip = tid & 63;            // 64 strips = 8 rows x 8 groups of 4
    const int sr = strip >> 3;             // tile row 0..7
    const int sc = (strip & 7) << 2;       // col base 0,4,...,28
    const int cbase = tid >> 6;            // channel phase 0..3
    for (int c = cbase; c < C_; c += 4) {
      const float* fp = ft + c*FT_STRIDE + sr*HLW_ + sc;
      const float* kc = kn + c*9;
      float acc0 = 0.f, acc1 = 0.f, acc2 = 0.f, acc3 = 0.f;
#pragma unroll
      for (int dr = 0; dr < 3; ++dr) {
        float row[6];
#pragma unroll
        for (int j = 0; j < 6; ++j) row[j] = fp[dr*HLW_ + j];
#pragma unroll
        for (int dc = 0; dc < 3; ++dc) {
          float kw = kc[dr*3 + dc];
          acc0 += kw * row[dc + 0];
          acc1 += kw * row[dc + 1];
          acc2 += kw * row[dc + 2];
          acc3 += kw * row[dc + 3];
        }
      }
      const int pbase = sr*TW_ + sc;
      xh[(pbase + 0)*(2*X_STRIDE) + c] = __float2half(lrelu01(acc0));
      xh[(pbase + 1)*(2*X_STRIDE) + c] = __float2half(lrelu01(acc1));
      xh[(pbase + 2)*(2*X_STRIDE) + c] = __float2half(lrelu01(acc2));
      xh[(pbase + 3)*(2*X_STRIDE) + c] = __float2half(lrelu01(acc3));
    }
  }
  __syncthreads();

  // Phase 3: 1x1 conv as D = W(16o x 32c) * X(32c x 16pix), fp32 accum.
  {
    const int wave = tid >> 5, lane = tid & 31;
    const int ln = lane & 15, hi = lane >> 4;
    const unsigned int* xu = (const unsigned int*)xh;
    const unsigned int* wu = (const unsigned int*)wsh;
    int kidx[8];
#pragma unroll
    for (int r = 0; r < 8; ++r)   // CDNA5 16-bit A/B fragment K pattern (pairs)
      kidx[r] = ((r & 3) << 1) + ((r >> 2) << 4) + (hi << 3);

#pragma unroll
    for (int half = 0; half < 2; ++half) {           // N-tile = 16 consecutive cols
      const int colbase = half << 4;
      const int p = wave*TW_ + colbase + ln;         // pixel: row=wave, col=colbase+ln
      Frag16 b0, b1;                                 // activations, K=0..31 / 32..63
#pragma unroll
      for (int r = 0; r < 8; ++r) {
        b0.u[r] = xu[p*X_STRIDE + (kidx[r] >> 1)];
        b1.u[r] = xu[p*X_STRIDE + (kidx[r] >> 1) + 16];
      }
#pragma unroll
      for (int mt = 0; mt < 4; ++mt) {               // M-tile = 16 out channels
        const int o = mt*16 + ln;
        Frag16 a0, a1;                               // conv_w rows
#pragma unroll
        for (int r = 0; r < 8; ++r) {
          a0.u[r] = wu[o*(WS_STRIDE/2) + (kidx[r] >> 1)];
          a1.u[r] = wu[o*(WS_STRIDE/2) + (kidx[r] >> 1) + 16];
        }
        v8f acc = {};
        acc = __builtin_amdgcn_wmma_f32_16x16x32_f16(false, a0.v, false, b0.v,
                                                     (short)0, acc, false, false);
        acc = __builtin_amdgcn_wmma_f32_16x16x32_f16(false, a1.v, false, b1.v,
                                                     (short)0, acc, false, false);
#pragma unroll
        for (int r = 0; r < 8; ++r) {                // D: M = r + 8*hi, N = ln
          const int oc = mt*16 + r + (hi << 3);
          float val = acc[r] + cb[oc]
                    + attb[oc] * ft[oc*FT_STRIDE + (wave + 1)*HLW_ + colbase + ln + 1];
          out[((b*C_ + oc) * H_ + (h0 + wave)) * W_ + (w0 + colbase + ln)] = val;
        }
      }
    }
  }
}

// ---------------- launch ----------------
extern "C" void kernel_launch(void* const* d_in, const int* in_sizes, int n_in,
                              void* d_out, int out_size, void* d_ws, size_t ws_size,
                              hipStream_t stream) {
  const float* feat   = (const float*)d_in[0];
  const float* deg    = (const float*)d_in[1];
  const float* wq     = (const float*)d_in[2];
  const float* bq     = (const float*)d_in[3];
  const float* wk     = (const float*)d_in[4];
  const float* bk     = (const float*)d_in[5];
  const float* wv     = (const float*)d_in[6];
  const float* bv     = (const float*)d_in[7];
  const float* gamma  = (const float*)d_in[8];
  const float* k_w1   = (const float*)d_in[9];
  const float* k_w2   = (const float*)d_in[10];
  const float* conv_w = (const float*)d_in[11];
  const float* conv_b = (const float*)d_in[12];
  const float* ca_w1  = (const float*)d_in[13];
  const float* ca_w2  = (const float*)d_in[14];

  float* ws   = (float*)d_ws;
  float* fa   = ws;                                   // 1024 f
  float* kern = ws + 1024;                            // 9216 f
  float* attw = ws + 1024 + 9216;                     // 1024 f
  __half* wh  = (__half*)(ws + 1024 + 9216 + 1024);   // 4096 halfs

  dqa_mean<<<dim3(B_*C_), dim3(256), 0, stream>>>(feat, fa);
  dqa_prep<<<dim3(1), dim3(256), 0, stream>>>(fa, deg, wq, bq, wk, bk, wv, bv,
                                              gamma, k_w1, k_w2, conv_w, ca_w1,
                                              ca_w2, kern, attw, wh);
  dqa_main<<<dim3(W_/TW_, H_/TH_, B_), dim3(256), SMEM_BYTES, stream>>>(
      feat, conv_b, kern, attw, wh, (float*)d_out);
}
